// DELTAModel_23974507446796
// MI455X (gfx1250) — matrix operand order repaired and checked by reading, not compiled, not run on hardware
//
#include <hip/hip_runtime.h>
#include <hip/hip_bf16.h>
#include <math.h>

#define NN 50000
#define EE 1600000
#define DD 64
#define DE 32
#define HH 4
#define LL 3

typedef __attribute__((ext_vector_type(16))) __bf16 v16bf;
typedef __attribute__((ext_vector_type(8)))  float  v8f;

union AF { v16bf v; unsigned u[8]; uint4 q[2]; };

static __device__ inline unsigned short f2bf(float f) {
    union { float f; unsigned u; } c; c.f = f;
    unsigned r = c.u + 0x7FFFu + ((c.u >> 16) & 1u);   // RNE
    return (unsigned short)(r >> 16);
}

static __device__ inline uint2 pk4bf(float4 v) {
    uint2 p;
    p.x = (unsigned)f2bf(v.x) | ((unsigned)f2bf(v.y) << 16);
    p.y = (unsigned)f2bf(v.z) | ((unsigned)f2bf(v.w) << 16);
    return p;
}

static __device__ inline unsigned fenc(float f) {      // order-preserving float->uint
    unsigned u = __float_as_uint(f);
    return (u & 0x80000000u) ? ~u : (u | 0x80000000u);
}
static __device__ inline float fdec(unsigned k) {
    unsigned u = (k & 0x80000000u) ? (k & 0x7FFFFFFFu) : ~k;
    return __uint_as_float(u);
}

static __device__ inline float gelu_tanh(float x) {
    float x3 = x * x * x;
    return 0.5f * x * (1.0f + tanhf(0.7978845608f * (x + 0.044715f * x3)));
}

// ---------------- utility kernels ----------------
__global__ void fill_u32_kernel(unsigned* __restrict__ p, unsigned val, int n) {
    int i = blockIdx.x * blockDim.x + threadIdx.x;
    if (i < n) p[i] = val;
}

// Pack W[L][K][NC] (f32, K = KC*32) into WMMA B-fragment layout:
// out[ ((layer*NT + t)*KC + c)*32*8 + lane*8 + j ] = bf16(W[kk][n]) | bf16(W[kk+1][n])<<16
// with n = t*16 + (lane&15), kk = c*32 + (lane>>4)*16 + 2j.
__global__ void pack_b_kernel(const float* __restrict__ W, unsigned* __restrict__ out,
                              int KC, int NT, int NC, int total) {
    int idx = blockIdx.x * blockDim.x + threadIdx.x;
    if (idx >= total) return;
    int j     = idx & 7;
    int lane  = (idx >> 3) & 31;
    int rest  = idx >> 8;
    int c     = rest % KC;
    int t     = (rest / KC) % NT;
    int layer = rest / (KC * NT);
    int n  = t * 16 + (lane & 15);
    int kk = c * 32 + ((lane >> 4) << 4) + (j << 1);
    const float* Wl = W + (size_t)layer * (size_t)(KC * 32) * NC;
    unsigned lo = f2bf(Wl[(size_t)kk * NC + n]);
    unsigned hi = f2bf(Wl[(size_t)(kk + 1) * NC + n]);
    out[idx] = lo | (hi << 16);
}

// ---------------- y[N,64] = x[N,64] @ W[64,64]  (bf16 WMMA, f32 accum) -------
__launch_bounds__(256)
__global__ void gemm64_kernel(const float* __restrict__ x,
                              const unsigned* __restrict__ wpk,   // packed B frags (this layer)
                              float* __restrict__ y, int ntiles) {
    __shared__ __align__(16) unsigned short sA[8][16 * 64];

    const int tid  = threadIdx.x;
    const int wave = tid >> 5;
    const int lane = tid & 31;
    const int m    = lane & 15;
    const int hgrp = lane >> 4;

    // B fragments: 4 N-tiles x 2 K-chunks, loaded once, coalesced b128 loads
    AF b[4][2];
    const uint4* wp = (const uint4*)wpk;
    #pragma unroll
    for (int t = 0; t < 4; ++t)
        #pragma unroll
        for (int c = 0; c < 2; ++c) {
            int fi = ((t * 2 + c) * 32 + lane) * 2;
            b[t][c].q[0] = wp[fi];
            b[t][c].q[1] = wp[fi + 1];
        }

    const int tilesPerPass = gridDim.x * 8;
    const int nIter = (ntiles + tilesPerPass - 1) / tilesPerPass;

    for (int it = 0; it < nIter; ++it) {
        const int tile = it * tilesPerPass + blockIdx.x * 8 + wave;
        const bool act = tile < ntiles;
        if (act) {
            const float4* xr = (const float4*)(x + (size_t)tile * 1024);
            unsigned short* st = sA[wave];
            #pragma unroll
            for (int i4 = 0; i4 < 8; ++i4) {
                int c4 = lane + i4 * 32;               // 0..255 float4 chunks
                *(uint2*)(st + c4 * 4) = pk4bf(xr[c4]);
            }
        }
        __syncthreads();
        if (act) {
            AF a[2];
            const unsigned short* row = sA[wave] + m * 64;
            #pragma unroll
            for (int c = 0; c < 2; ++c) {
                a[c].q[0] = *(const uint4*)(row + c * 32 + hgrp * 16);
                a[c].q[1] = *(const uint4*)(row + c * 32 + 8 + hgrp * 16);
            }
            float* yr = y + (size_t)tile * 16 * 64;
            #pragma unroll
            for (int t = 0; t < 4; ++t) {
                v8f acc = {};
                #pragma unroll
                for (int c = 0; c < 2; ++c)
                    acc = __builtin_amdgcn_wmma_f32_16x16x32_bf16(
                        false, a[c].v, false, b[t][c].v, (short)0, acc, false, false);
                #pragma unroll
                for (int r = 0; r < 8; ++r)
                    yr[(r + 8 * hgrp) * 64 + t * 16 + m] = acc[r];
            }
        }
        __syncthreads();
    }
}

// ---------------- per-edge attention scores + segment max ----------------
__global__ void score_kernel(const float* __restrict__ q, const float* __restrict__ k,
                             const float* __restrict__ e, const int* __restrict__ srcv,
                             const int* __restrict__ dstv, const float* __restrict__ We,
                             float* __restrict__ sc_p, unsigned* __restrict__ mseg) {
    int ed = blockIdx.x * blockDim.x + threadIdx.x;
    if (ed >= EE) return;
    int s = srcv[ed], d = dstv[ed];
    const float4* qd = (const float4*)(q + (size_t)d * 64);
    const float4* ks = (const float4*)(k + (size_t)s * 64);
    float sc[4];
    #pragma unroll
    for (int h = 0; h < 4; ++h) {
        float acc = 0.f;
        #pragma unroll
        for (int j = 0; j < 4; ++j) {
            float4 a = qd[h * 4 + j], b = ks[h * 4 + j];
            acc += a.x * b.x + a.y * b.y + a.z * b.z + a.w * b.w;
        }
        sc[h] = acc * 0.25f;                      // 1/sqrt(16)
    }
    const float4* er = (const float4*)(e + (size_t)ed * 32);
    #pragma unroll
    for (int j4 = 0; j4 < 8; ++j4) {
        float4 ev = er[j4];
        const float* wr = We + j4 * 16;
        #pragma unroll
        for (int h = 0; h < 4; ++h)
            sc[h] += ev.x * wr[h] + ev.y * wr[4 + h] + ev.z * wr[8 + h] + ev.w * wr[12 + h];
    }
    #pragma unroll
    for (int h = 0; h < 4; ++h) {
        sc_p[(size_t)ed * 4 + h] = sc[h];
        atomicMax(&mseg[(size_t)d * 4 + h], fenc(sc[h]));
    }
}

// ---------------- exp(score - max) and segment sum ----------------
__global__ void expsum_kernel(float* __restrict__ sc_p, const unsigned* __restrict__ mseg,
                              float* __restrict__ denom, const int* __restrict__ dstv) {
    int ed = blockIdx.x * blockDim.x + threadIdx.x;
    if (ed >= EE) return;
    int d = dstv[ed];
    #pragma unroll
    for (int h = 0; h < 4; ++h) {
        float mx = fdec(mseg[(size_t)d * 4 + h]);
        float p = expf(sc_p[(size_t)ed * 4 + h] - mx);
        sc_p[(size_t)ed * 4 + h] = p;
        atomicAdd(&denom[(size_t)d * 4 + h], p);
    }
}

// ---------------- alpha * v[src] scatter-add ----------------
__global__ void agg_kernel(const float* __restrict__ p, const float* __restrict__ denom,
                           const float* __restrict__ v, const int* __restrict__ srcv,
                           const int* __restrict__ dstv, float* __restrict__ agg) {
    int ed = blockIdx.x * blockDim.x + threadIdx.x;
    if (ed >= EE) return;
    int s = srcv[ed], d = dstv[ed];
    float al[4];
    #pragma unroll
    for (int h = 0; h < 4; ++h)
        al[h] = p[(size_t)ed * 4 + h] / (denom[(size_t)d * 4 + h] + 1e-9f);
    const float4* vr = (const float4*)(v + (size_t)s * 64);
    float* ar = agg + (size_t)d * 64;
    #pragma unroll
    for (int j4 = 0; j4 < 16; ++j4) {
        float4 vv = vr[j4];
        float a0 = al[j4 >> 2];
        atomicAdd(&ar[j4 * 4 + 0], a0 * vv.x);
        atomicAdd(&ar[j4 * 4 + 1], a0 * vv.y);
        atomicAdd(&ar[j4 * 4 + 2], a0 * vv.z);
        atomicAdd(&ar[j4 * 4 + 3], a0 * vv.w);
    }
}

// ---------------- node residual + LayerNorm (in place) ----------------
__global__ void ln_node_kernel(float* __restrict__ h, const float* __restrict__ t,
                               const float* __restrict__ g, const float* __restrict__ b) {
    int r = blockIdx.x * blockDim.x + threadIdx.x;
    if (r >= NN) return;
    float4* hr = (float4*)(h + (size_t)r * 64);
    const float4* tr = (const float4*)(t + (size_t)r * 64);
    float x[64]; float mu = 0.f;
    #pragma unroll
    for (int j4 = 0; j4 < 16; ++j4) {
        float4 a = hr[j4], c = tr[j4];
        x[j4 * 4 + 0] = a.x + c.x; x[j4 * 4 + 1] = a.y + c.y;
        x[j4 * 4 + 2] = a.z + c.z; x[j4 * 4 + 3] = a.w + c.w;
        mu += x[j4 * 4 + 0] + x[j4 * 4 + 1] + x[j4 * 4 + 2] + x[j4 * 4 + 3];
    }
    mu *= (1.f / 64.f);
    float var = 0.f;
    #pragma unroll
    for (int j = 0; j < 64; ++j) { float d0 = x[j] - mu; var += d0 * d0; }
    var *= (1.f / 64.f);
    float rs = rsqrtf(var + 1e-5f);
    #pragma unroll
    for (int j4 = 0; j4 < 16; ++j4) {
        float4 o;
        o.x = (x[j4 * 4 + 0] - mu) * rs * g[j4 * 4 + 0] + b[j4 * 4 + 0];
        o.y = (x[j4 * 4 + 1] - mu) * rs * g[j4 * 4 + 1] + b[j4 * 4 + 1];
        o.z = (x[j4 * 4 + 2] - mu) * rs * g[j4 * 4 + 2] + b[j4 * 4 + 2];
        o.w = (x[j4 * 4 + 3] - mu) * rs * g[j4 * 4 + 3] + b[j4 * 4 + 3];
        hr[j4] = o;
    }
}

// ---------------- edge MLP: LN(e + gelu([h[src]|h[dst]|e] @ Wem + bem)) -----
__launch_bounds__(256)
__global__ void edge_mlp_kernel(const float* __restrict__ h, float* __restrict__ e,
                                const int* __restrict__ srcv, const int* __restrict__ dstv,
                                const unsigned* __restrict__ wpk,   // packed Wem frags (layer)
                                const float* __restrict__ bem,
                                const float* __restrict__ ge, const float* __restrict__ be,
                                int ntiles) {
    __shared__ __align__(16) unsigned short sA[8][16 * 160];  // also reused as f32 [16][32]

    const int tid  = threadIdx.x;
    const int wave = tid >> 5;
    const int lane = tid & 31;
    const int m    = lane & 15;
    const int hgrp = lane >> 4;

    // B fragments: 5 K-chunks x 2 N-tiles, loaded once from packed global, kept in regs
    AF bfrag[5][2];
    const uint4* wp = (const uint4*)wpk;
    #pragma unroll
    for (int t = 0; t < 2; ++t)
        #pragma unroll
        for (int c = 0; c < 5; ++c) {
            int fi = ((t * 5 + c) * 32 + lane) * 2;
            bfrag[c][t].q[0] = wp[fi];
            bfrag[c][t].q[1] = wp[fi + 1];
        }

    const int tilesPerPass = gridDim.x * 8;
    const int nIter = (ntiles + tilesPerPass - 1) / tilesPerPass;

    for (int it = 0; it < nIter; ++it) {
        const int tile = it * tilesPerPass + blockIdx.x * 8 + wave;
        const bool act = tile < ntiles;
        const long e0 = (long)tile * 16;
        if (act) {
            unsigned short* st = sA[wave];
            for (int mm = 0; mm < 16; ++mm) {
                long edg = e0 + mm;                       // wave-uniform -> scalar loads
                const float4* hs = (const float4*)(h + (size_t)srcv[edg] * 64);
                const float4* hd = (const float4*)(h + (size_t)dstv[edg] * 64);
                const float4* ee = (const float4*)(e + (size_t)edg * 32);
                // row = 40 float4 chunks: 0..15 h[src], 16..31 h[dst], 32..39 e
                {
                    int c4 = lane;                        // 0..31, always < 40
                    float4 val = (c4 < 16) ? hs[c4] : hd[c4 - 16];
                    *(uint2*)(st + mm * 160 + c4 * 4) = pk4bf(val);
                }
                if (lane < 8) {
                    int c4 = lane + 32;                   // 32..39 -> e chunk
                    *(uint2*)(st + mm * 160 + c4 * 4) = pk4bf(ee[lane]);
                }
            }
        }
        __syncthreads();
        if (act) {
            v8f acc0 = {}; v8f acc1 = {};
            const unsigned short* row = sA[wave] + m * 160;
            #pragma unroll
            for (int c = 0; c < 5; ++c) {
                AF a;
                a.q[0] = *(const uint4*)(row + c * 32 + hgrp * 16);
                a.q[1] = *(const uint4*)(row + c * 32 + 8 + hgrp * 16);
                acc0 = __builtin_amdgcn_wmma_f32_16x16x32_bf16(
                    false, a.v, false, bfrag[c][0].v, (short)0, acc0, false, false);
                acc1 = __builtin_amdgcn_wmma_f32_16x16x32_bf16(
                    false, a.v, false, bfrag[c][1].v, (short)0, acc1, false, false);
            }
            // bias + gelu, drop into LDS (overlay staging buffer, reads already consumed)
            float* g = (float*)sA[wave];
            #pragma unroll
            for (int r = 0; r < 8; ++r) {
                int row2 = r + 8 * hgrp;
                g[row2 * 32 + m]      = gelu_tanh(acc0[r] + bem[m]);
                g[row2 * 32 + 16 + m] = gelu_tanh(acc1[r] + bem[16 + m]);
            }
        }
        __syncthreads();
        if (act && lane < 16) {
            long edg = e0 + lane;
            float4* er = (float4*)(e + (size_t)edg * 32);
            const float4* g4 = (const float4*)((const float*)sA[wave] + lane * 32);
            float x[32]; float mu = 0.f;
            #pragma unroll
            for (int j4 = 0; j4 < 8; ++j4) {
                float4 a = er[j4], c = g4[j4];
                x[j4 * 4 + 0] = a.x + c.x; x[j4 * 4 + 1] = a.y + c.y;
                x[j4 * 4 + 2] = a.z + c.z; x[j4 * 4 + 3] = a.w + c.w;
                mu += x[j4 * 4 + 0] + x[j4 * 4 + 1] + x[j4 * 4 + 2] + x[j4 * 4 + 3];
            }
            mu *= (1.f / 32.f);
            float var = 0.f;
            #pragma unroll
            for (int j = 0; j < 32; ++j) { float d0 = x[j] - mu; var += d0 * d0; }
            var *= (1.f / 32.f);
            float rs = rsqrtf(var + 1e-5f);
            #pragma unroll
            for (int j4 = 0; j4 < 8; ++j4) {
                float4 o;
                o.x = (x[j4 * 4 + 0] - mu) * rs * ge[j4 * 4 + 0] + be[j4 * 4 + 0];
                o.y = (x[j4 * 4 + 1] - mu) * rs * ge[j4 * 4 + 1] + be[j4 * 4 + 1];
                o.z = (x[j4 * 4 + 2] - mu) * rs * ge[j4 * 4 + 2] + be[j4 * 4 + 2];
                o.w = (x[j4 * 4 + 3] - mu) * rs * ge[j4 * 4 + 3] + be[j4 * 4 + 3];
                er[j4] = o;
            }
        }
        __syncthreads();
    }
}

// ---------------- host side ----------------
extern "C" void kernel_launch(void* const* d_in, const int* in_sizes, int n_in,
                              void* d_out, int out_size, void* d_ws, size_t ws_size,
                              hipStream_t stream) {
    (void)in_sizes; (void)n_in; (void)out_size; (void)ws_size;

    const float* node_f = (const float*)d_in[0];
    const float* edge_f = (const float*)d_in[1];
    const int*   eidx   = (const int*)d_in[2];
    const float* Wq     = (const float*)d_in[3];
    const float* Wk     = (const float*)d_in[4];
    const float* Wv     = (const float*)d_in[5];
    const float* Wo     = (const float*)d_in[6];
    const float* We     = (const float*)d_in[7];
    const float* Wem    = (const float*)d_in[8];
    const float* bem    = (const float*)d_in[9];
    const float* gn     = (const float*)d_in[10];
    const float* bn     = (const float*)d_in[11];
    const float* ge     = (const float*)d_in[12];
    const float* be     = (const float*)d_in[13];
    const int* srcv = eidx;
    const int* dstv = eidx + EE;

    char* base = (char*)d_ws;
    size_t off = 0;
    auto alloc = [&](size_t bytes) -> char* {
        char* p = base + off;
        off += (bytes + 255) & ~(size_t)255;
        return p;
    };

    float* h     = (float*)alloc((size_t)NN * 64 * 4);
    float* e     = (float*)alloc((size_t)EE * 32 * 4);
    float* q     = (float*)alloc((size_t)NN * 64 * 4);
    float* k     = (float*)alloc((size_t)NN * 64 * 4);
    float* v     = (float*)alloc((size_t)NN * 64 * 4);
    float* tbuf  = (float*)alloc((size_t)NN * 64 * 4);
    float* agg   = (float*)alloc((size_t)NN * 64 * 4);
    float* sc_p  = (float*)alloc((size_t)EE * 4 * 4);
    unsigned* mseg  = (unsigned*)alloc((size_t)NN * 4 * 4);
    float*    denom = (float*)   alloc((size_t)NN * 4 * 4);
    // packed B fragments: per layer, gemm = 4*2*256 dwords, wem = 2*5*256 dwords
    const int GPK = 4 * 2 * 256;   // 2048 dwords / layer
    const int EPK = 2 * 5 * 256;   // 2560 dwords / layer
    unsigned* wq_pk  = (unsigned*)alloc((size_t)LL * GPK * 4);
    unsigned* wk_pk  = (unsigned*)alloc((size_t)LL * GPK * 4);
    unsigned* wv_pk  = (unsigned*)alloc((size_t)LL * GPK * 4);
    unsigned* wo_pk  = (unsigned*)alloc((size_t)LL * GPK * 4);
    unsigned* wem_pk = (unsigned*)alloc((size_t)LL * EPK * 4);

    const int CB = 256;
    pack_b_kernel<<<(LL * GPK + CB - 1) / CB, CB, 0, stream>>>(Wq,  wq_pk,  2, 4, 64, LL * GPK);
    pack_b_kernel<<<(LL * GPK + CB - 1) / CB, CB, 0, stream>>>(Wk,  wk_pk,  2, 4, 64, LL * GPK);
    pack_b_kernel<<<(LL * GPK + CB - 1) / CB, CB, 0, stream>>>(Wv,  wv_pk,  2, 4, 64, LL * GPK);
    pack_b_kernel<<<(LL * GPK + CB - 1) / CB, CB, 0, stream>>>(Wo,  wo_pk,  2, 4, 64, LL * GPK);
    pack_b_kernel<<<(LL * EPK + CB - 1) / CB, CB, 0, stream>>>(Wem, wem_pk, 5, 2, 32, LL * EPK);

    hipMemcpyAsync(h, node_f, (size_t)NN * 64 * 4, hipMemcpyDeviceToDevice, stream);
    hipMemcpyAsync(e, edge_f, (size_t)EE * 32 * 4, hipMemcpyDeviceToDevice, stream);

    const int nodeTiles = NN / 16;   // 3125, exact
    const int edgeTiles = EE / 16;   // 100000, exact
    const int eGrid = (EE + CB - 1) / CB;

    for (int i = 0; i < LL; ++i) {
        gemm64_kernel<<<200, 256, 0, stream>>>(h, wq_pk + (size_t)i * GPK, q, nodeTiles);
        gemm64_kernel<<<200, 256, 0, stream>>>(h, wk_pk + (size_t)i * GPK, k, nodeTiles);
        gemm64_kernel<<<200, 256, 0, stream>>>(h, wv_pk + (size_t)i * GPK, v, nodeTiles);

        fill_u32_kernel<<<(NN * 4 + CB - 1) / CB, CB, 0, stream>>>(mseg, 0x007FFFFFu, NN * 4); // enc(-inf)
        hipMemsetAsync(denom, 0, (size_t)NN * 4 * 4, stream);
        hipMemsetAsync(agg,   0, (size_t)NN * 64 * 4, stream);

        score_kernel <<<eGrid, CB, 0, stream>>>(q, k, e, srcv, dstv, We + (size_t)i * 128, sc_p, mseg);
        expsum_kernel<<<eGrid, CB, 0, stream>>>(sc_p, mseg, denom, dstv);
        agg_kernel   <<<eGrid, CB, 0, stream>>>(sc_p, denom, v, srcv, dstv, agg);

        gemm64_kernel<<<200, 256, 0, stream>>>(agg, wo_pk + (size_t)i * GPK, tbuf, nodeTiles);
        ln_node_kernel<<<(NN + CB - 1) / CB, CB, 0, stream>>>(h, tbuf, gn + (size_t)i * 64, bn + (size_t)i * 64);

        edge_mlp_kernel<<<1024, 256, 0, stream>>>(h, e, srcv, dstv,
                                                  wem_pk + (size_t)i * EPK,
                                                  bem + (size_t)i * 32,
                                                  ge + (size_t)i * 32, be + (size_t)i * 32,
                                                  edgeTiles);
    }

    hipMemcpyAsync(d_out, h, (size_t)NN * 64 * 4, hipMemcpyDeviceToDevice, stream);
}
